// NeighborhoodSelfAttentionModule_65000035058176
// MI455X (gfx1250) — matrix operand order repaired and checked by reading, not compiled, not run on hardware
//
#include <hip/hip_runtime.h>
#include <hip/hip_bf16.h>

typedef __attribute__((ext_vector_type(16))) __bf16 v16bf;
typedef __attribute__((ext_vector_type(8)))  __bf16 v8bf;   // 16 B, 16-B aligned
typedef __attribute__((ext_vector_type(8)))  float  v8f;

#define HH 32
#define WW 32
#define ZZ 16
#define PP (HH * WW * ZZ)   // 16384 positions
#define CC 128              // channels
#define HEADS 4
#define HD 32               // head dim
#define QKV_SEC ((size_t)HEADS * PP * HD)   // floats per q/k/v section
#define LDP (CC + 8)        // padded LDS row (halves): 272 B rows, 16-B aligned

// Load one A fragment (16x32 bf16, contiguous k order matching B) from a row-major fp32
// matrix: lane m = lane&15, K elements [0..15]->kb..kb+15.
__device__ __forceinline__ v16bf load_a_frag(const float* __restrict__ arow,
                                             int kb) {
    v16bf a;
#pragma unroll
    for (int i = 0; i < 16; ++i)
        a[i] = (__bf16)arow[i + kb];
    return a;
}

// Load one B fragment (32x16 bf16) from the K-contiguous LDS tile:
// lane n = lane&15 -> row, K half-offset 0/16 by lane hi bit; two b128 loads.
__device__ __forceinline__ v16bf load_b_frag(const __bf16* __restrict__ bptr) {
    v8bf lo = *(const v8bf*)(bptr);
    v8bf hi = *(const v8bf*)(bptr + 8);
    v16bf b;
#pragma unroll
    for (int i = 0; i < 8; ++i) { b[i] = lo[i]; b[i + 8] = hi[i]; }
    return b;
}

// ---------------------------------------------------------------------------
// Kernel A: QKV projection GEMM.  qkv[o][p] = sum_c Wqkv[o][c]*x[c][p] + b[o]
// 64-position x-tile staged K-contiguous in LDS; each wave holds 4 A-frags
// (K=128 resident) and sweeps 4 N-subtiles -> 16 WMMAs, 4x A reuse.
// Fuses bias, q-scale, heads-last relayout; vectorized b128 writeback.
// ---------------------------------------------------------------------------
__global__ __launch_bounds__(256) void qkv_gemm_kernel(
    const float* __restrict__ x,      // [C][P]
    const float* __restrict__ Wqkv,   // [3C][C]
    const float* __restrict__ bqkv,   // [3C]
    float* __restrict__ qkv_ws)       // q | k | v, each [HEADS][P][HD]
{
    __shared__ __bf16 sB[64 * LDP];   // x tile: [pos 0..63][K 0..127], padded

    const int lane = threadIdx.x & 31;
    const int wave = threadIdx.x >> 5;
    const int p0   = blockIdx.x * 64;

    // cooperative stage: coalesced global rows -> K-contiguous LDS rows
    for (int t = threadIdx.x; t < 64 * CC; t += 256) {
        int j = t & 63, c = t >> 6;
        sB[j * LDP + c] = (__bf16)x[(size_t)c * PP + p0 + j];
    }
    __syncthreads();

    const int ct = blockIdx.y * 8 + wave;   // channel tile 0..23 (384 outputs)
    const int o0 = ct * 16;
    const int m  = lane & 15;
    const int hi = lane >> 4;
    const int kb = hi << 4;                 // A frag K half-offset 0/16
    const int n  = lane & 15;
    const int k0 = hi << 4;                 // B frag K half-offset 0/16

    // all 4 A fragments resident (K = 0..127)
    v16bf afrag[4];
    const float* arow = Wqkv + (size_t)(o0 + m) * CC;
    __builtin_prefetch(arow, 0, 1);
#pragma unroll
    for (int kk = 0; kk < 4; ++kk)
        afrag[kk] = load_a_frag(arow + kk * 32, kb);

    v8f acc[4];
#pragma unroll
    for (int ns = 0; ns < 4; ++ns) { v8f z = {}; acc[ns] = z; }

#pragma unroll
    for (int ns = 0; ns < 4; ++ns) {
        const __bf16* brow = &sB[(ns * 16 + n) * LDP];
#pragma unroll
        for (int kk = 0; kk < 4; ++kk) {
            v16bf b = load_b_frag(brow + kk * 32 + k0);
            acc[ns] = __builtin_amdgcn_wmma_f32_16x16x32_bf16(
                false, afrag[kk], false, b, (short)0, acc[ns], false, false);
        }
    }

    // writeback: tile never straddles a head (o0 16-aligned, head block 32),
    // so each lane's 8 values are contiguous head-dims at one position.
    const float scale = 0.17677669529663688f;       // 32^-0.5
    const int sec  = o0 >> 7;                       // 0=q 1=k 2=v
    const int head = (o0 & 127) >> 5;
    const int d0   = (o0 & 31) + hi * 8;
#pragma unroll
    for (int ns = 0; ns < 4; ++ns) {
        int p = p0 + ns * 16 + n;
        float vals[8];
#pragma unroll
        for (int r = 0; r < 8; ++r) {
            float v = acc[ns][r] + bqkv[o0 + hi * 8 + r];
            vals[r] = (sec == 0) ? v * scale : v;
        }
        float* dst = qkv_ws + (size_t)sec * QKV_SEC +
                     (size_t)head * PP * HD + (size_t)p * HD + d0;
        *(float4*)(dst)     = make_float4(vals[0], vals[1], vals[2], vals[3]);
        *(float4*)(dst + 4) = make_float4(vals[4], vals[5], vals[6], vals[7]);
    }
}

// ---------------------------------------------------------------------------
// Kernel B: neighborhood attention. One thread per (head, position).
// 27 neighbors x 32-dim dots, rpb bias, stable softmax, weighted-v sum.
// All k/v reads are 32 contiguous floats (L2-resident).
// ---------------------------------------------------------------------------
__global__ __launch_bounds__(256) void nat_attn_kernel(
    const float* __restrict__ qkv_ws,
    const float* __restrict__ rpb,     // [4][5][5][5]
    float* __restrict__ attn_out)      // [C][P]
{
    int tid  = blockIdx.x * 256 + threadIdx.x;      // HEADS*PP threads
    int head = tid >> 14;
    int p    = tid & (PP - 1);
    int h    = p >> 9;          // / (W*Z)
    int w    = (p >> 4) & 31;
    int z    = p & 15;

    const float* qb    = qkv_ws + (size_t)head * PP * HD + (size_t)p * HD;
    const float* kbase = qkv_ws + QKV_SEC     + (size_t)head * PP * HD;
    const float* vbase = qkv_ws + 2 * QKV_SEC + (size_t)head * PP * HD;

    float q[HD];
#pragma unroll
    for (int d = 0; d < HD; ++d) q[d] = qb[d];

    // window starts: clip(pos-1, 0, L-3)
    int sh = min(max(h - 1, 0), HH - 3);
    int sw = min(max(w - 1, 0), WW - 3);
    int sz = min(max(z - 1, 0), ZZ - 3);

    float logit[27];
    float mx = -1e30f;
#pragma unroll
    for (int dh = 0; dh < 3; ++dh)
#pragma unroll
    for (int dw = 0; dw < 3; ++dw)
#pragma unroll
    for (int dz = 0; dz < 3; ++dz) {
        int nh = sh + dh, nw = sw + dw, nz = sz + dz;
        int np = (nh << 9) + (nw << 4) + nz;
        const float* kp = kbase + (size_t)np * HD;
        float dot = 0.f;
#pragma unroll
        for (int d = 0; d < HD; ++d) dot += q[d] * kp[d];
        int rh = nh - h + 2, rw = nw - w + 2, rz = nz - z + 2;
        dot += rpb[((head * 5 + rh) * 5 + rw) * 5 + rz];
        logit[(dh * 3 + dw) * 3 + dz] = dot;
        mx = fmaxf(mx, dot);
    }

    float sum = 0.f;
#pragma unroll
    for (int j = 0; j < 27; ++j) { logit[j] = __expf(logit[j] - mx); sum += logit[j]; }
    float inv = 1.f / sum;

    float out[HD];
#pragma unroll
    for (int d = 0; d < HD; ++d) out[d] = 0.f;
#pragma unroll
    for (int dh = 0; dh < 3; ++dh)
#pragma unroll
    for (int dw = 0; dw < 3; ++dw)
#pragma unroll
    for (int dz = 0; dz < 3; ++dz) {
        int np = ((sh + dh) << 9) + ((sw + dw) << 4) + (sz + dz);
        const float* vp = vbase + (size_t)np * HD;
        float wgt = logit[(dh * 3 + dw) * 3 + dz] * inv;
#pragma unroll
        for (int d = 0; d < HD; ++d) out[d] += wgt * vp[d];
    }

    // channel-major writeback so the projection GEMM's B-loads are contiguous
#pragma unroll
    for (int d = 0; d < HD; ++d)
        attn_out[(size_t)(head * HD + d) * PP + p] = out[d];
}

// ---------------------------------------------------------------------------
// Kernel C: output projection GEMM.  out[o][p] = sum_c Wp[o][c]*attn[c][p]+bp[o]
// Same 64-position tile / resident-A / 16-WMMA structure as kernel A.
// Writes final (B,C,H,W,Z) layout directly.
// ---------------------------------------------------------------------------
__global__ __launch_bounds__(256) void proj_gemm_kernel(
    const float* __restrict__ attn,   // [C][P]
    const float* __restrict__ Wp,     // [C][C]
    const float* __restrict__ bp,     // [C]
    float* __restrict__ out)          // [C][P]
{
    __shared__ __bf16 sB[64 * LDP];

    const int lane = threadIdx.x & 31;
    const int wave = threadIdx.x >> 5;
    const int p0   = blockIdx.x * 64;

    for (int t = threadIdx.x; t < 64 * CC; t += 256) {
        int j = t & 63, c = t >> 6;
        sB[j * LDP + c] = (__bf16)attn[(size_t)c * PP + p0 + j];
    }
    __syncthreads();

    const int o0 = wave * 16;   // 8 waves cover all 128 output channels
    const int m  = lane & 15;
    const int hi = lane >> 4;
    const int kb = hi << 4;
    const int n  = lane & 15;
    const int k0 = hi << 4;

    v16bf afrag[4];
    const float* arow = Wp + (size_t)(o0 + m) * CC;
    __builtin_prefetch(arow, 0, 1);
#pragma unroll
    for (int kk = 0; kk < 4; ++kk)
        afrag[kk] = load_a_frag(arow + kk * 32, kb);

    v8f acc[4];
#pragma unroll
    for (int ns = 0; ns < 4; ++ns) { v8f z = {}; acc[ns] = z; }

#pragma unroll
    for (int ns = 0; ns < 4; ++ns) {
        const __bf16* brow = &sB[(ns * 16 + n) * LDP];
#pragma unroll
        for (int kk = 0; kk < 4; ++kk) {
            v16bf b = load_b_frag(brow + kk * 32 + k0);
            acc[ns] = __builtin_amdgcn_wmma_f32_16x16x32_bf16(
                false, afrag[kk], false, b, (short)0, acc[ns], false, false);
        }
    }

#pragma unroll
    for (int ns = 0; ns < 4; ++ns) {
#pragma unroll
        for (int r = 0; r < 8; ++r) {
            int o = o0 + r + hi * 8;
            int p = p0 + ns * 16 + n;
            out[(size_t)o * PP + p] = acc[ns][r] + bp[o];
        }
    }
}

extern "C" void kernel_launch(void* const* d_in, const int* in_sizes, int n_in,
                              void* d_out, int out_size, void* d_ws, size_t ws_size,
                              hipStream_t stream) {
    const float* x    = (const float*)d_in[0];
    const float* Wqkv = (const float*)d_in[1];
    const float* bqkv = (const float*)d_in[2];
    const float* rpb  = (const float*)d_in[3];
    const float* Wp   = (const float*)d_in[4];
    const float* bp   = (const float*)d_in[5];

    float* ws       = (float*)d_ws;
    float* qkv_ws   = ws;                        // 3 * 4*16384*32 floats = 24 MB
    float* attn_out = ws + 3 * QKV_SEC;          // 128*16384 floats      =  8 MB

    // QKV projection: 256 position-tiles x 3 channel-tile groups
    qkv_gemm_kernel<<<dim3(PP / 64, 3), dim3(256), 0, stream>>>(x, Wqkv, bqkv, qkv_ws);

    // Attention: one thread per (head, position)
    nat_attn_kernel<<<dim3(HEADS * PP / 256), dim3(256), 0, stream>>>(qkv_ws, rpb, attn_out);

    // Output projection
    proj_gemm_kernel<<<dim3(PP / 64), dim3(256), 0, stream>>>(attn_out, Wp, bp, (float*)d_out);
}
